// MultiHeadSelfAttention_54296976556687
// MI455X (gfx1250) — compile-verified, matrix operand order checked by reading, not run on hardware
//
#include <hip/hip_runtime.h>
#include <hip/hip_bf16.h>
#include <math.h>

// ---------------------------------------------------------------------------
// Transformer encoder layer for MI455X (gfx1250, wave32, WMMA 16x16x32 f16).
// B=4, S=2048, D=1024, H=16, HD=64, FF=2048.
// GEMMs: LDS double-buffered K=64 tiles filled with async global->LDS copies
// (ASYNCcnt), consumed by v_wmma_f32_16x16x32_f16.  Attention is flash-style.
// ---------------------------------------------------------------------------

#define BB   4
#define SS   2048
#define DD   1024
#define HH   16
#define HDIM 64
#define FF2  2048
#define MTOT (BB * SS)   // 8192 rows of activations

typedef __attribute__((ext_vector_type(16))) _Float16 v16h;
typedef __attribute__((ext_vector_type(8)))  _Float16 v8h;
typedef __attribute__((ext_vector_type(8)))  float    v8f;
typedef __attribute__((__vector_size__(16))) int      v4i;

union FragU { v16h f; v8h h[2]; };

#ifndef __has_builtin
#define __has_builtin(x) 0
#endif

#if __has_builtin(__builtin_amdgcn_global_load_async_to_lds_b128)
#define HAVE_ASYNC_LDS 1
#else
#define HAVE_ASYNC_LDS 0
#endif

#if HAVE_ASYNC_LDS
#if __has_builtin(__builtin_amdgcn_s_wait_asynccnt)
#define ASYNC_WAIT(n) __builtin_amdgcn_s_wait_asynccnt(n)
#else
#define ASYNC_WAIT(n) asm volatile("s_wait_asynccnt %0" ::"n"(n) : "memory")
#endif
#else
#define ASYNC_WAIT(n) do {} while (0)
#endif

// 16-byte global -> LDS copy (async on gfx1250; signature confirmed by hipcc
// diagnostics: v4i* in AS1, v4i* in AS3, imm offset, imm cpol).
__device__ __forceinline__ void copy16_g2l(const _Float16* g, _Float16* l) {
#if HAVE_ASYNC_LDS
    __builtin_amdgcn_global_load_async_to_lds_b128(
        (__attribute__((address_space(1))) v4i*)(uintptr_t)g,
        (__attribute__((address_space(3))) v4i*)(uint32_t)(uintptr_t)l,
        0, 0);
#else
    *(v8h*)l = *(const v8h*)g;
#endif
}

// ---------------------------------------------------------------------------
// WMMA wrapper (probe-verified signature, 8 args)
// ---------------------------------------------------------------------------
__device__ __forceinline__ v8f wmma_f16(v16h a, v16h b, v8f c) {
    return __builtin_amdgcn_wmma_f32_16x16x32_f16(
        /*neg_a=*/false, a, /*neg_b=*/false, b,
        /*c_mod=*/(short)0, c, /*reuse_a=*/false, /*reuse_b=*/false);
}

// ---------------------------------------------------------------------------
// Fragment loader for A (16xK tile, M per lane) or B (Kx16 tile from a
// transposed N-major buffer).  Per ISA 16-bit A layout:
//   lane L: m = L%16, h = L/16
//   elements 0..7  -> K = kb + h*8 + e
//   elements 8..15 -> K = kb + 16 + h*8 + (e-8)
// Both halves are 16B-contiguous -> two b128 loads (global or ds).
// ---------------------------------------------------------------------------
__device__ __forceinline__ v16h load_frag(const _Float16* base, int ld, int kb) {
    int lane = threadIdx.x & 31;
    int i = lane & 15;
    int h = lane >> 4;
    const _Float16* p = base + (size_t)i * ld + kb + h * 8;
    FragU u;
    u.h[0] = *(const v8h*)(p);
    u.h[1] = *(const v8h*)(p + 16);
    return u.f;
}

// ---------------------------------------------------------------------------
// Weight conversion: W (K x N, fp32, N contiguous) -> Wt (N x K, f16, K contig)
// ---------------------------------------------------------------------------
__global__ void convert_wT(const float* __restrict__ W, _Float16* __restrict__ Wt,
                           int K, int N) {
    size_t idx = (size_t)blockIdx.x * blockDim.x + threadIdx.x;
    size_t total = (size_t)K * N;
    if (idx >= total) return;
    int n = (int)(idx / K);
    int k = (int)(idx % K);
    Wt[idx] = (_Float16)W[(size_t)k * N + n];
}

// ---------------------------------------------------------------------------
// LayerNorm: fp32 row (D=1024) -> f16 normalized row. One block per row.
// ---------------------------------------------------------------------------
__global__ __launch_bounds__(256) void ln_f16(const float* __restrict__ x,
                                              const float* __restrict__ g,
                                              const float* __restrict__ b,
                                              _Float16* __restrict__ out) {
    size_t row = blockIdx.x;
    int t = threadIdx.x;
    const float4* xr = (const float4*)(x + row * DD);
    float4 v = xr[t];
    float s  = v.x + v.y + v.z + v.w;
    float s2 = v.x * v.x + v.y * v.y + v.z * v.z + v.w * v.w;

    __shared__ float r1[256];
    __shared__ float r2[256];
    r1[t] = s; r2[t] = s2;
    __syncthreads();
    for (int o = 128; o > 0; o >>= 1) {
        if (t < o) { r1[t] += r1[t + o]; r2[t] += r2[t + o]; }
        __syncthreads();
    }
    float mu  = r1[0] * (1.0f / DD);
    float var = r2[0] * (1.0f / DD) - mu * mu;
    float rsd = rsqrtf(var + 1e-5f);

    float4 gv = ((const float4*)g)[t];
    float4 bv = ((const float4*)b)[t];
    _Float16* o4 = out + row * DD + (size_t)t * 4;
    o4[0] = (_Float16)((v.x - mu) * rsd * gv.x + bv.x);
    o4[1] = (_Float16)((v.y - mu) * rsd * gv.y + bv.y);
    o4[2] = (_Float16)((v.z - mu) * rsd * gv.z + bv.z);
    o4[3] = (_Float16)((v.w - mu) * rsd * gv.w + bv.w);
}

// ---------------------------------------------------------------------------
// Tiled WMMA GEMM:  C[M,N] = A[M,K](f16) @ Bt[N,K](f16)^T  (+ epilogue)
// Block = 256 threads = 8 wave32s arranged 2(M) x 4(N); wave tile 32x32;
// block tile 64 x 128.  K-steps of 64 are double-buffered in LDS:
//   fill(next tile) = 6 async global->LDS b128 copies per thread;
//   s_wait_asynccnt 6 keeps the next fill fully in flight while the current
//   tile is guaranteed resident (async loads complete in order);
//   then 8x v_wmma_f32_16x16x32_f16 per wave from LDS between barriers.
// Row pitch 72 halves = 144B = 36-bank stride (gcd(36,64)=4 -> the 16 lanes
// of each half-wave hit distinct banks on ds_load_b128).
// ---------------------------------------------------------------------------
enum { EPI_H16 = 0, EPI_VT = 1, EPI_F32RES = 2, EPI_GELU = 3 };

#define KSTEP  64
#define APITCH 72

__device__ __forceinline__ void fill_tiles(
    const _Float16* __restrict__ A, const _Float16* __restrict__ Bt, int K,
    int rowBase, int colBase, int k0,
    _Float16 (*Ash)[APITCH], _Float16 (*Bsh)[APITCH]) {
    int t = threadIdx.x;
#pragma unroll
    for (int j = 0; j < 2; ++j) {  // A tile: 64 rows x 8 chunks = 512, 2/thread
        int idx = t + j * 256;
        int row = idx >> 3, c = (idx & 7) * 8;
        copy16_g2l(A + (size_t)(rowBase + row) * K + k0 + c, &Ash[row][c]);
    }
#pragma unroll
    for (int j = 0; j < 4; ++j) {  // B tile: 128 rows x 8 chunks = 1024, 4/thread
        int idx = t + j * 256;
        int row = idx >> 3, c = (idx & 7) * 8;
        copy16_g2l(Bt + (size_t)(colBase + row) * K + k0 + c, &Bsh[row][c]);
    }
}

template <int EPI>
__global__ __launch_bounds__(256) void gemm_wmma(
    const _Float16* __restrict__ A, const _Float16* __restrict__ Bt,
    const float* __restrict__ bias, const float* __restrict__ res,
    float* __restrict__ outF, _Float16* __restrict__ outH,
    int M, int N, int K) {
    int lane = threadIdx.x & 31;
    int wave = threadIdx.x >> 5;
    int wm = wave & 1;
    int wn = wave >> 1;
    int rowBase = blockIdx.y * 64;
    int colBase = blockIdx.x * 128;
    int row0 = rowBase + wm * 32;
    int col0 = colBase + wn * 32;

    __shared__ __align__(16) _Float16 Ash[2][64][APITCH];
    __shared__ __align__(16) _Float16 Bsh[2][128][APITCH];

    v8f acc[2][2] = {};

    fill_tiles(A, Bt, K, rowBase, colBase, 0, Ash[0], Bsh[0]);

    int buf = 0;
    for (int k0 = 0; k0 < K; k0 += KSTEP) {
        bool more = (k0 + KSTEP) < K;
        if (more)
            fill_tiles(A, Bt, K, rowBase, colBase, k0 + KSTEP,
                       Ash[buf ^ 1], Bsh[buf ^ 1]);
#if HAVE_ASYNC_LDS
        if (more) ASYNC_WAIT(6); else ASYNC_WAIT(0);
#endif
        __syncthreads();

#pragma unroll
        for (int kk = 0; kk < KSTEP; kk += 32) {
            v16h a0 = load_frag(&Ash[buf][wm * 32][0],      APITCH, kk);
            v16h a1 = load_frag(&Ash[buf][wm * 32 + 16][0], APITCH, kk);
            v16h b0 = load_frag(&Bsh[buf][wn * 32][0],      APITCH, kk);
            v16h b1 = load_frag(&Bsh[buf][wn * 32 + 16][0], APITCH, kk);
            acc[0][0] = wmma_f16(a0, b0, acc[0][0]);
            acc[0][1] = wmma_f16(a0, b1, acc[0][1]);
            acc[1][0] = wmma_f16(a1, b0, acc[1][0]);
            acc[1][1] = wmma_f16(a1, b1, acc[1][1]);
        }

        __syncthreads();
        buf ^= 1;
    }

    int i = lane & 15;
    int h = lane >> 4;
#pragma unroll
    for (int mt = 0; mt < 2; ++mt) {
#pragma unroll
        for (int nt = 0; nt < 2; ++nt) {
            int col = col0 + nt * 16 + i;
            float bv = bias[col];
            int m0 = row0 + mt * 16 + 8 * h;   // rows m0..m0+7 (element r)
            if (EPI == EPI_VT) {
                // store f16 transposed: [B, H, HD, S], s contiguous -> b128
                int bidx = m0 / SS;
                int s0   = m0 % SS;
                int head = col >> 6;          // col / HD
                int d    = col & 63;          // col % HD
                _Float16* dst = outH +
                    ((((size_t)bidx * HH + head) * HDIM + d) * SS + s0);
                v8h pk;
#pragma unroll
                for (int r = 0; r < 8; ++r)
                    pk[r] = (_Float16)(acc[mt][nt][r] + bv);
                *(v8h*)dst = pk;
            } else if (EPI == EPI_H16) {
#pragma unroll
                for (int r = 0; r < 8; ++r)
                    outH[(size_t)(m0 + r) * N + col] =
                        (_Float16)(acc[mt][nt][r] + bv);
            } else if (EPI == EPI_F32RES) {
#pragma unroll
                for (int r = 0; r < 8; ++r) {
                    size_t idx = (size_t)(m0 + r) * N + col;
                    outF[idx] = acc[mt][nt][r] + bv + res[idx];
                }
            } else { // EPI_GELU  (exact gelu: 0.5*x*(1+erf(x/sqrt(2))))
#pragma unroll
                for (int r = 0; r < 8; ++r) {
                    float t = acc[mt][nt][r] + bv;
                    float gl = 0.5f * t * (1.0f + erff(t * 0.70710678118f));
                    outH[(size_t)(m0 + r) * N + col] = (_Float16)gl;
                }
            }
        }
    }
}

// ---------------------------------------------------------------------------
// Flash attention, one wave per block.  Block handles 16 queries of one
// (batch, head); streams keys in tiles of 32 with online softmax.
// q,k row-major f16 [B,S,D]; v transposed f16 [B,H,HD,S]; ctx f16 [B,S,D].
// ---------------------------------------------------------------------------
__global__ __launch_bounds__(32) void attn_flash(
    const _Float16* __restrict__ q, const _Float16* __restrict__ k,
    const _Float16* __restrict__ vT, _Float16* __restrict__ ctx) {
    int lane = threadIdx.x & 31;
    int i = lane & 15;
    int h = lane >> 4;
    int q0   = blockIdx.x * 16;
    int head = blockIdx.y;
    int b    = blockIdx.z;

    const _Float16* qbase = q + ((size_t)(b * SS + q0)) * DD + head * HDIM;
    v16h qf0 = load_frag(qbase, DD, 0);    // d = 0..31
    v16h qf1 = load_frag(qbase, DD, 32);   // d = 32..63

    v8f acc[4] = {};
    float mrow[8], lrow[8];
#pragma unroll
    for (int r = 0; r < 8; ++r) { mrow[r] = -3.0e38f; lrow[r] = 0.0f; }

    __shared__ __align__(16) _Float16 P[16 * 32];

    const _Float16* vhead = vT + ((size_t)(b * HH + head)) * HDIM * SS;

    for (int kt = 0; kt < SS; kt += 32) {
        // ---- scores: 16 queries x 32 keys -------------------------------
        v8f s0 = {}, s1 = {};
        const _Float16* kb0 = k + ((size_t)(b * SS + kt)) * DD + head * HDIM;
        const _Float16* kb1 = kb0 + (size_t)16 * DD;
        s0 = wmma_f16(qf0, load_frag(kb0, DD, 0),  s0);
        s0 = wmma_f16(qf1, load_frag(kb0, DD, 32), s0);
        s1 = wmma_f16(qf0, load_frag(kb1, DD, 0),  s1);
        s1 = wmma_f16(qf1, load_frag(kb1, DD, 32), s1);
        s0 = s0 * 0.125f;   // HD^-0.5
        s1 = s1 * 0.125f;

        // ---- online softmax (row = element r, per half-wave) ------------
        float t[8];
#pragma unroll
        for (int r = 0; r < 8; ++r) t[r] = fmaxf(s0[r], s1[r]);
#pragma unroll
        for (int mask = 1; mask <= 8; mask <<= 1)
#pragma unroll
            for (int r = 0; r < 8; ++r)
                t[r] = fmaxf(t[r], __shfl_xor(t[r], mask, 32));

        float p0[8], p1[8], rs[8];
#pragma unroll
        for (int r = 0; r < 8; ++r) {
            float mn = fmaxf(mrow[r], t[r]);
            float fs = __expf(mrow[r] - mn);
            mrow[r] = mn;
            p0[r] = __expf(s0[r] - mn);
            p1[r] = __expf(s1[r] - mn);
            rs[r] = p0[r] + p1[r];
#pragma unroll
            for (int dt = 0; dt < 4; ++dt) acc[dt][r] *= fs;
            lrow[r] *= fs;
        }
#pragma unroll
        for (int mask = 1; mask <= 8; mask <<= 1)
#pragma unroll
            for (int r = 0; r < 8; ++r)
                rs[r] += __shfl_xor(rs[r], mask, 32);
#pragma unroll
        for (int r = 0; r < 8; ++r) lrow[r] += rs[r];

        // ---- D-layout -> A-layout via LDS -------------------------------
#pragma unroll
        for (int r = 0; r < 8; ++r) {
            P[(r + 8 * h) * 32 + i]      = (_Float16)p0[r];
            P[(r + 8 * h) * 32 + 16 + i] = (_Float16)p1[r];
        }
        __syncthreads();
        v16h pa = load_frag(P, 32, 0);   // 16 queries x 32 keys
        __syncthreads();

        // ---- ctx += P @ V  (V^T: fixed d per lane, keys contiguous) -----
#pragma unroll
        for (int dt = 0; dt < 4; ++dt) {
            const _Float16* vb = vhead + (size_t)(dt * 16) * SS;
            acc[dt] = wmma_f16(pa, load_frag(vb, SS, kt), acc[dt]);
        }
    }

    // ---- normalize and store f16 row-major [B,S,D] ----------------------
#pragma unroll
    for (int dt = 0; dt < 4; ++dt) {
#pragma unroll
        for (int r = 0; r < 8; ++r) {
            float o = acc[dt][r] / lrow[r];
            int m = q0 + 8 * h + r;
            ctx[((size_t)(b * SS + m)) * DD + head * HDIM + dt * 16 + i] =
                (_Float16)o;
        }
    }
}

// ---------------------------------------------------------------------------
// Launcher
// ---------------------------------------------------------------------------
extern "C" void kernel_launch(void* const* d_in, const int* in_sizes, int n_in,
                              void* d_out, int out_size, void* d_ws, size_t ws_size,
                              hipStream_t stream) {
    const float* x   = (const float*)d_in[0];
    const float* Wq  = (const float*)d_in[1];
    const float* bq  = (const float*)d_in[2];
    const float* Wk  = (const float*)d_in[3];
    const float* bk  = (const float*)d_in[4];
    const float* Wv  = (const float*)d_in[5];
    const float* bv  = (const float*)d_in[6];
    const float* Wo  = (const float*)d_in[7];
    const float* bo  = (const float*)d_in[8];
    const float* g1  = (const float*)d_in[9];
    const float* b1  = (const float*)d_in[10];
    const float* g2  = (const float*)d_in[11];
    const float* b2  = (const float*)d_in[12];
    const float* W1  = (const float*)d_in[13];
    const float* bf1 = (const float*)d_in[14];
    const float* W2  = (const float*)d_in[15];
    const float* bf2 = (const float*)d_in[16];
    float* out = (float*)d_out;

    char* ws = (char*)d_ws;
    size_t off = 0;
    auto alloc = [&](size_t bytes) {
        char* p = ws + off;
        off += (bytes + 255) & ~(size_t)255;
        return p;
    };
    _Float16* xn   = (_Float16*)alloc((size_t)MTOT * DD * 2);
    _Float16* qb   = (_Float16*)alloc((size_t)MTOT * DD * 2);
    _Float16* kb   = (_Float16*)alloc((size_t)MTOT * DD * 2);
    _Float16* vTb  = (_Float16*)alloc((size_t)MTOT * DD * 2);
    _Float16* ctx  = (_Float16*)alloc((size_t)MTOT * DD * 2);
    float*    x2   = (float*)   alloc((size_t)MTOT * DD * 4);
    _Float16* xn2  = (_Float16*)alloc((size_t)MTOT * DD * 2);
    _Float16* hbuf = (_Float16*)alloc((size_t)MTOT * FF2 * 2);
    _Float16* WqT  = (_Float16*)alloc((size_t)DD * DD * 2);
    _Float16* WkT  = (_Float16*)alloc((size_t)DD * DD * 2);
    _Float16* WvT  = (_Float16*)alloc((size_t)DD * DD * 2);
    _Float16* WoT  = (_Float16*)alloc((size_t)DD * DD * 2);
    _Float16* W1T  = (_Float16*)alloc((size_t)DD * FF2 * 2);
    _Float16* W2T  = (_Float16*)alloc((size_t)FF2 * DD * 2);
    if (off > ws_size) return;

    // 1) convert weights to f16, transposed (N-major, K-contiguous)
    {
        int n1 = DD * DD, n2 = DD * FF2;
        convert_wT<<<(n1 + 255) / 256, 256, 0, stream>>>(Wq, WqT, DD, DD);
        convert_wT<<<(n1 + 255) / 256, 256, 0, stream>>>(Wk, WkT, DD, DD);
        convert_wT<<<(n1 + 255) / 256, 256, 0, stream>>>(Wv, WvT, DD, DD);
        convert_wT<<<(n1 + 255) / 256, 256, 0, stream>>>(Wo, WoT, DD, DD);
        convert_wT<<<(n2 + 255) / 256, 256, 0, stream>>>(W1, W1T, DD, FF2);
        convert_wT<<<(n2 + 255) / 256, 256, 0, stream>>>(W2, W2T, FF2, DD);
    }

    // 2) LN1
    ln_f16<<<MTOT, 256, 0, stream>>>(x, g1, b1, xn);

    // 3) QKV projections
    dim3 gD(DD / 128, MTOT / 64);
    gemm_wmma<EPI_H16><<<gD, 256, 0, stream>>>(xn, WqT, bq, nullptr, nullptr, qb,  MTOT, DD, DD);
    gemm_wmma<EPI_H16><<<gD, 256, 0, stream>>>(xn, WkT, bk, nullptr, nullptr, kb,  MTOT, DD, DD);
    gemm_wmma<EPI_VT> <<<gD, 256, 0, stream>>>(xn, WvT, bv, nullptr, nullptr, vTb, MTOT, DD, DD);

    // 4) flash attention
    attn_flash<<<dim3(SS / 16, HH, BB), 32, 0, stream>>>(qb, kb, vTb, ctx);

    // 5) out projection + residual (fp32)
    gemm_wmma<EPI_F32RES><<<gD, 256, 0, stream>>>(ctx, WoT, bo, x, x2, nullptr, MTOT, DD, DD);

    // 6) LN2
    ln_f16<<<MTOT, 256, 0, stream>>>(x2, g2, b2, xn2);

    // 7) FFN1 + exact GELU
    dim3 gF(FF2 / 128, MTOT / 64);
    gemm_wmma<EPI_GELU><<<gF, 256, 0, stream>>>(xn2, W1T, bf1, nullptr, nullptr, hbuf, MTOT, FF2, DD);

    // 8) FFN2 + residual -> fp32 output
    gemm_wmma<EPI_F32RES><<<gD, 256, 0, stream>>>(hbuf, W2T, bf2, x2, out, nullptr, MTOT, DD, FF2);
}
